// model_15796889715473
// MI455X (gfx1250) — compile-verified, matrix operand order checked by reading, not run on hardware
//
#include <hip/hip_runtime.h>
#include <hip/hip_bf16.h>

// ---------------------------------------------------------------------------
// CDNA5 (gfx1250) implementation of the NNConv GNN forward.
// Edge-MLP linears run on v_wmma_f32_16x16x32_f16 (16 edges x 16 outs, K=32).
// BatchNorm (training mode) handled with two-pass stats (atomic sum/sumsq).
// ---------------------------------------------------------------------------

typedef __attribute__((ext_vector_type(16))) _Float16 v16h;
typedef __attribute__((ext_vector_type(8)))  float    v8f;

#define EPS 1e-5f

// Hardware fp32 atomic add (global_atomic_add_f32), no CAS loop.
static __device__ inline void atomAdd(float* p, float v) {
  unsafeAtomicAdd(p, v);
}

// ------------------------------ device helpers -----------------------------

struct ConvP {
  const float *bias, *bn0_b, *bn0_g, *bn1_b, *bn1_g, *lb, *lw;
  const float *a_bnb, *a_bng, *a_l1b, *a_l1w, *a_l2w;
  const float *b1, *b2, *b3, *c1, *c2;
  const float *n1b, *n1g, *n2b, *n2g, *n3b, *n3g;
  const float *w1, *w2, *w3, *root;
};

struct EdgeArgs {
  ConvP P;
  const float* ea;        // [E,11]
  const float* eaStats;   // mean[11], rstd[11]
  const float* bnBuf;     // 5 x [scale[32] shift[32]]
  float*       raw;       // 5 x [sum[32] sumsq[32]]
  const int*   src;
  const int*   dst;
  const float* xin;       // [N,5]
  float*       agg;       // [N,5]
  int E;
};

struct ESh {                       // per-wave LDS tile set
  float h0[512], X1[512], X2[512], T0[512], T1[512], B[1024];
};

// Zero-padded [32][32] weight tile from a row-major [KI][NO] fp32 matrix.
static __device__ inline void wave_fill_B(float* B, const float* W, int KI,
                                          int NO, int lane) {
  for (int idx = lane; idx < 32 * 32; idx += 32) {
    int k = idx >> 5, n = idx & 31;
    B[idx] = (k < KI && n < NO) ? W[k * NO + n] : 0.f;
  }
}

// One 16x16x32 WMMA: A = LDS [16][32] fp32 (M,K), B = LDS [32][32] fp32,
// N-tile base nb. Fragment assembly follows ISA 7.12.2 16-bit layouts:
// lane L: M/N = L&15, hb = L>>4;  half j -> K = 16*(j/8) + 8*hb + (j&7).
static __device__ inline v8f wave_wmma(const float* A, const float* B, int nb,
                                       int lane) {
  int r = lane & 15, hb = lane >> 4;
  v16h a, b;
#pragma unroll
  for (int j = 0; j < 16; ++j) {
    int k = ((j >> 3) << 4) + (hb << 3) + (j & 7);
    a[j] = (_Float16)A[r * 32 + k];
    b[j] = (_Float16)B[k * 32 + nb + r];
  }
  v8f c = {};
  return __builtin_amdgcn_wmma_f32_16x16x32_f16(false, a, false, b, (short)0,
                                                c, false, false);
}

// Store accumulator to LDS [16][32] (+bias), zeroing cols >= NO for padding.
// col is fixed per lane, so load the bias once (avoids branchy per-row loads).
static __device__ inline void wave_store(float* O, v8f c, int nb, int NO,
                                         const float* bias, int lane) {
  int n = lane & 15, hb = lane >> 4;
  int col = nb + n;
  bool valid = col < NO;
  float bv = (valid && bias) ? bias[col] : 0.f;
#pragma unroll
  for (int r = 0; r < 8; ++r) {
    int m = r + 8 * hb;
    O[m * 32 + col] = valid ? (c[r] + bv) : 0.f;
  }
}

// O[16][NO] = A[16][KI] @ W[KI][NO] + bias   (KI<=32, NO<=32)
static __device__ inline void wave_lin(const float* A, float* O,
                                       const float* W, const float* bias,
                                       int KI, int NO, float* B, int lane) {
  wave_fill_B(B, W, KI, NO, lane);
  __syncthreads();
  v8f c0 = wave_wmma(A, B, 0, lane);
  v8f c1 = {};
  if (NO > 16) c1 = wave_wmma(A, B, 16, lane);
  wave_store(O, c0, 0, NO, bias, lane);
  if (NO > 16) {
    wave_store(O, c1, 16, NO, bias, lane);
  } else {
    int n = lane & 15, hb = lane >> 4;
#pragma unroll
    for (int r = 0; r < 8; ++r) O[(r + 8 * hb) * 32 + 16 + n] = 0.f;
  }
  __syncthreads();
}

// Column sum / sumsq over valid rows of an LDS tile -> atomic into raw[64].
static __device__ inline void wave_stats(const float* O, int NO, float* raw,
                                         int nvalid, int lane) {
  if (lane < NO) {
    float s = 0.f, q = 0.f;
    for (int m = 0; m < nvalid; ++m) {
      float v = O[m * 32 + lane];
      s += v; q += v * v;
    }
    atomAdd(&raw[lane], s);
    atomAdd(&raw[32 + lane], q);
  }
}

// dst = relu(scale*src + shift) per column (cols >= NO zeroed)
static __device__ inline void wave_bnrelu(float* dst, const float* src,
                                          const float* ss, int NO, int lane) {
  for (int idx = lane; idx < 512; idx += 32) {
    int c = idx & 31;
    dst[idx] = (c < NO) ? fmaxf(0.f, ss[c] * src[idx] + ss[32 + c]) : 0.f;
  }
}

// ------------------------------ edge kernel --------------------------------
// STAGE 1: stats of lin1 output (n1) + att lin1 output (att bn)
// STAGE 2: stats of lin2 output (n2)
// STAGE 3: stats of lin3 output (n3)
// STAGE 5: stats of final 25-wide output (bn1)
// STAGE 6: message pass (gather x[src], 5x5 matvec, scatter-add to dst)
template <int STAGE>
__global__ __launch_bounds__(128) void edge_kernel(EdgeArgs a) {
  __shared__ ESh sm[4];
  const int lane = threadIdx.x & 31, wave = threadIdx.x >> 5;
  ESh& S = sm[wave];
  const int base = (blockIdx.x * 4 + wave) * 16;
  const int nvalid = max(0, min(16, a.E - base));
  const ConvP& P = a.P;

  // h0 = BN0(edge_attr)
  for (int idx = lane; idx < 512; idx += 32) {
    int m = idx >> 5, k = idx & 31, e = base + m;
    float v = 0.f;
    if (k < 11 && e < a.E) {
      float t = (a.ea[e * 11 + k] - a.eaStats[k]) * a.eaStats[11 + k];
      v = P.bn0_g[k] * t + P.bn0_b[k];
    }
    S.h0[idx] = v;
  }

  wave_lin(S.h0, S.T0, P.w1, P.b1, 11, 18, S.B, lane);      // lin1
  if (STAGE == 1) {
    wave_stats(S.T0, 18, a.raw + 0 * 64, nvalid, lane);
    wave_lin(S.h0, S.T1, P.a_l1w, P.a_l1b, 11, 6, S.B, lane);  // att lin1
    wave_stats(S.T1, 6, a.raw + 3 * 64, nvalid, lane);
    return;
  }
  wave_bnrelu(S.X1, S.T0, a.bnBuf + 0 * 64, 18, lane);       // x1
  __syncthreads();

  wave_lin(S.X1, S.T0, P.w2, P.b2, 18, 18, S.B, lane);       // lin2
  if (STAGE == 2) { wave_stats(S.T0, 18, a.raw + 1 * 64, nvalid, lane); return; }
  wave_bnrelu(S.X2, S.T0, a.bnBuf + 1 * 64, 18, lane);       // x2
  __syncthreads();

  const float c10 = P.c1[0], c11 = P.c1[1];
  for (int idx = lane; idx < 512; idx += 32)
    S.T0[idx] = c10 * S.X1[idx] + c11 * S.X2[idx];           // conv2d(2,1)
  __syncthreads();

  wave_lin(S.T0, S.T1, P.w3, P.b3, 18, 18, S.B, lane);       // lin3
  if (STAGE == 3) { wave_stats(S.T1, 18, a.raw + 2 * 64, nvalid, lane); return; }

  const float c20 = P.c2[0], c21 = P.c2[1], c22 = P.c2[2];
  const float* n3ss = a.bnBuf + 2 * 64;
  for (int idx = lane; idx < 512; idx += 32) {               // conv2d(3,1)
    int c = idx & 31;
    float x3 = (c < 18) ? fmaxf(0.f, n3ss[c] * S.T1[idx] + n3ss[32 + c]) : 0.f;
    S.T0[idx] = c20 * S.X1[idx] + c21 * S.X2[idx] + c22 * x3;
  }
  __syncthreads();

  // attention gate: hardsigmoid(relu(bn(h0@l1w+l1b)) @ l2w)
  wave_lin(S.h0, S.T1, P.a_l1w, P.a_l1b, 11, 6, S.B, lane);
  wave_bnrelu(S.T1, S.T1, a.bnBuf + 3 * 64, 6, lane);
  __syncthreads();
  wave_lin(S.T1, S.X1, P.a_l2w, nullptr, 6, 18, S.B, lane);
  for (int idx = lane; idx < 512; idx += 32) {
    float f = fminf(fmaxf(S.X1[idx] * (1.f / 6.f) + 0.5f, 0.f), 1.f);
    S.T0[idx] *= f;
  }
  __syncthreads();

  wave_lin(S.T0, S.T1, P.lw, P.lb, 18, 25, S.B, lane);       // 18->25
  if (STAGE == 5) { wave_stats(S.T1, 25, a.raw + 4 * 64, nvalid, lane); return; }

  // STAGE 6: apply bn1 then message pass
  const float* b1ss = a.bnBuf + 4 * 64;
  for (int idx = lane; idx < 512; idx += 32) {
    int c = idx & 31;
    if (c < 25) S.T1[idx] = b1ss[c] * S.T1[idx] + b1ss[32 + c];
  }
  __syncthreads();
  if (lane < 16 && lane < nvalid) {
    int e = base + lane, s = a.src[e], d = a.dst[e];
    float xs[5];
#pragma unroll
    for (int j = 0; j < 5; ++j) xs[j] = a.xin[s * 5 + j];
#pragma unroll
    for (int o = 0; o < 5; ++o) {
      float m = 0.f;
#pragma unroll
      for (int j = 0; j < 5; ++j) m += xs[j] * S.T1[lane * 32 + j * 5 + o];
      atomAdd(&a.agg[d * 5 + o], m);
    }
  }
}

// ------------------------------ small kernels ------------------------------

__global__ void ea_stats_kernel(const float* ea, float* raw, int E) {
  float s[11], q[11];
  for (int k = 0; k < 11; ++k) { s[k] = 0.f; q[k] = 0.f; }
  for (int e = blockIdx.x * blockDim.x + threadIdx.x; e < E;
       e += gridDim.x * blockDim.x)
    for (int k = 0; k < 11; ++k) {
      float v = ea[e * 11 + k];
      s[k] += v; q[k] += v * v;
    }
  for (int k = 0; k < 11; ++k) {
    atomAdd(&raw[k], s[k]);
    atomAdd(&raw[32 + k], q[k]);
  }
}

__global__ void ea_finalize_kernel(const float* raw, float* st, int E) {
  int t = threadIdx.x;
  if (t < 11) {
    float m = raw[t] / E, v = raw[32 + t] / E - m * m;
    st[t] = m; st[11 + t] = rsqrtf(v + EPS);
  }
}

__global__ void bn_finalize_kernel(const float* raw, const float* g,
                                   const float* b, float* ss, int n,
                                   float cnt) {
  int t = threadIdx.x;
  if (t < n) {
    float m = raw[t] / cnt, v = raw[32 + t] / cnt - m * m;
    float sc = g[t] * rsqrtf(v + EPS);
    ss[t] = sc; ss[32 + t] = b[t] - m * sc;
  }
}

__global__ void node_stats_kernel(const float* x, float* raw, int N) {
  int i = blockIdx.x * blockDim.x + threadIdx.x;
  if (i < N)
    for (int o = 0; o < 5; ++o) {
      float v = x[i * 5 + o];
      atomAdd(&raw[o], v);
      atomAdd(&raw[32 + o], v * v);
    }
}

// y = agg + x@root + bias (in place into agg), accumulate stats
__global__ void node_conv_kernel(const float* xin, float* agg,
                                 const float* root, const float* bias,
                                 float* raw, int N) {
  int i = blockIdx.x * blockDim.x + threadIdx.x;
  if (i < N) {
    float x[5], y[5];
#pragma unroll
    for (int j = 0; j < 5; ++j) x[j] = xin[i * 5 + j];
#pragma unroll
    for (int o = 0; o < 5; ++o) {
      float v = agg[i * 5 + o] + bias[o];
#pragma unroll
      for (int j = 0; j < 5; ++j) v += x[j] * root[j * 5 + o];
      y[o] = v;
    }
#pragma unroll
    for (int o = 0; o < 5; ++o) {
      agg[i * 5 + o] = y[o];
      atomAdd(&raw[o], y[o]);
      atomAdd(&raw[32 + o], y[o] * y[o]);
    }
  }
}

__global__ void node_apply_kernel(const float* y, const float* ss, float* out,
                                  const float* res, int n5, int doClamp) {
  int i = blockIdx.x * blockDim.x + threadIdx.x;
  if (i < n5) {
    int c = i % 5;
    float v = ss[c] * y[i] + ss[32 + c];
    if (doClamp) v = fminf(fmaxf(v, -1.f), 10.f);
    if (res) v += res[i];
    out[i] = v;
  }
}

__global__ void add3_kernel(float* out, const float* a, const float* b,
                            const float* c, int n) {
  int i = blockIdx.x * blockDim.x + threadIdx.x;
  if (i < n) out[i] = a[i] + b[i] + c[i];
}

struct Cat18 { const float* p[18]; };
__global__ void concat_kernel(Cat18 c, float* hcat, int N) {
  int i = blockIdx.x * blockDim.x + threadIdx.x;
  if (i < N)
    for (int t = 0; t < 18; ++t)
      for (int o = 0; o < 5; ++o)
        hcat[i * 90 + t * 5 + o] = c.p[t][i * 5 + o];
}

__global__ void catt1_kernel(const float* hcat, const float* w, const float* b,
                             float* t4, float* raw, int N) {
  int i = blockIdx.x * blockDim.x + threadIdx.x;
  if (i < N) {
    float acc[4];
    for (int j = 0; j < 4; ++j) acc[j] = b[j];
    for (int k = 0; k < 90; ++k) {
      float h = hcat[i * 90 + k];
      for (int j = 0; j < 4; ++j) acc[j] += h * w[k * 4 + j];
    }
    for (int j = 0; j < 4; ++j) {
      t4[i * 4 + j] = acc[j];
      atomAdd(&raw[j], acc[j]);
      atomAdd(&raw[32 + j], acc[j] * acc[j]);
    }
  }
}

__global__ void catt2_kernel(const float* hcat, const float* t4,
                             const float* ss, const float* l2w, float* outsum,
                             int N, int accum) {
  int i = blockIdx.x * blockDim.x + threadIdx.x;
  if (i < N) {
    float f4[4];
    for (int j = 0; j < 4; ++j)
      f4[j] = fmaxf(0.f, ss[j] * t4[i * 4 + j] + ss[32 + j]);
    for (int k = 0; k < 90; ++k) {
      float f = 0.f;
      for (int j = 0; j < 4; ++j) f += f4[j] * l2w[j * 90 + k];
      f = fminf(fmaxf(f, 0.f), 0.9f);
      float v = hcat[i * 90 + k] * f;
      outsum[i * 90 + k] = accum ? outsum[i * 90 + k] + v : v;
    }
  }
}

__global__ void final_lin_kernel(const float* h, const float* w,
                                 const float* b, float* out, int N) {
  int i = blockIdx.x * blockDim.x + threadIdx.x;
  if (i < N) {
    float v = b[0];
    for (int k = 0; k < 90; ++k) v += h[i * 90 + k] * w[k];
    out[i] = v;
  }
}

// ------------------------------ host side ----------------------------------
// Param leaves follow jax sorted-key flatten order:
//   d_in: 0=x 1=edge_index 2=edge_attr, then params:
//   att1(5) att2(5) blocks[18](87 each) bn(2) l19b l19w
// block: c1(27) c2(27) c3(27) n1.b n1.g n2.b n2.g n3.b n3.g
// conv:  bias, enn{bn0.b bn0.g bn1.b bn1.g lb lw,
//        ne{att{bn.b bn.g l1b l1w l2w} b1 b2 b3 c1 c2 n1.b n1.g n2.b n2.g
//           n3.b n3.g w1 w2 w3}}, root

static ConvP make_conv(void* const* d_in, int b, int c) {
  const int base = 3 + 10 + 87 * b + 27 * c;
  auto F = [&](int off) { return (const float*)d_in[base + off]; };
  ConvP P;
  P.bias = F(0);  P.bn0_b = F(1);  P.bn0_g = F(2);  P.bn1_b = F(3);
  P.bn1_g = F(4); P.lb = F(5);     P.lw = F(6);
  P.a_bnb = F(7); P.a_bng = F(8);  P.a_l1b = F(9);  P.a_l1w = F(10);
  P.a_l2w = F(11);
  P.b1 = F(12); P.b2 = F(13); P.b3 = F(14); P.c1 = F(15); P.c2 = F(16);
  P.n1b = F(17); P.n1g = F(18); P.n2b = F(19); P.n2g = F(20);
  P.n3b = F(21); P.n3g = F(22);
  P.w1 = F(23); P.w2 = F(24); P.w3 = F(25); P.root = F(26);
  return P;
}

struct Ctx {
  hipStream_t stream;
  int E, N;
  const float* ea;
  const int *src, *dst;
  float *eaStats, *eaRaw, *rawAll, *bnBuf, *nraw, *nbn, *araw, *abn, *agg;
};

static void run_conv(const Ctx& c, const ConvP& P, const float* xin) {
  const int eg = (c.E + 63) / 64, TB = 128;
  (void)hipMemsetAsync(c.rawAll, 0, 5 * 64 * sizeof(float), c.stream);
  EdgeArgs a{P, c.ea, c.eaStats, c.bnBuf, c.rawAll, c.src, c.dst, xin, c.agg,
             c.E};
  edge_kernel<1><<<eg, TB, 0, c.stream>>>(a);
  bn_finalize_kernel<<<1, 32, 0, c.stream>>>(c.rawAll, P.n1g, P.n1b, c.bnBuf,
                                             18, (float)c.E);
  bn_finalize_kernel<<<1, 32, 0, c.stream>>>(c.rawAll + 3 * 64, P.a_bng,
                                             P.a_bnb, c.bnBuf + 3 * 64, 6,
                                             (float)c.E);
  edge_kernel<2><<<eg, TB, 0, c.stream>>>(a);
  bn_finalize_kernel<<<1, 32, 0, c.stream>>>(c.rawAll + 64, P.n2g, P.n2b,
                                             c.bnBuf + 64, 18, (float)c.E);
  edge_kernel<3><<<eg, TB, 0, c.stream>>>(a);
  bn_finalize_kernel<<<1, 32, 0, c.stream>>>(c.rawAll + 2 * 64, P.n3g, P.n3b,
                                             c.bnBuf + 2 * 64, 18, (float)c.E);
  edge_kernel<5><<<eg, TB, 0, c.stream>>>(a);
  bn_finalize_kernel<<<1, 32, 0, c.stream>>>(c.rawAll + 4 * 64, P.bn1_g,
                                             P.bn1_b, c.bnBuf + 4 * 64, 25,
                                             (float)c.E);
  (void)hipMemsetAsync(c.agg, 0, (size_t)c.N * 5 * sizeof(float), c.stream);
  edge_kernel<6><<<eg, TB, 0, c.stream>>>(a);
  (void)hipMemsetAsync(c.nraw, 0, 64 * sizeof(float), c.stream);
  node_conv_kernel<<<(c.N + 255) / 256, 256, 0, c.stream>>>(
      xin, c.agg, P.root, P.bias, c.nraw, c.N);
}

static void run_block(const Ctx& c, void* const* d_in, int b, const float* xin,
                      float* out, float* wA, float* wB) {
  const int BB = 3 + 10 + 87 * b;
  auto F = [&](int off) { return (const float*)d_in[BB + off]; };
  const int n5 = c.N * 5, ng = (n5 + 255) / 256;
  ConvP p1 = make_conv(d_in, b, 0), p2 = make_conv(d_in, b, 1),
        p3 = make_conv(d_in, b, 2);
  run_conv(c, p1, xin);
  bn_finalize_kernel<<<1, 32, 0, c.stream>>>(c.nraw, F(82), F(81), c.nbn, 5,
                                             (float)c.N);
  node_apply_kernel<<<ng, 256, 0, c.stream>>>(c.agg, c.nbn, wA, nullptr, n5, 1);
  run_conv(c, p2, wA);
  bn_finalize_kernel<<<1, 32, 0, c.stream>>>(c.nraw, F(84), F(83), c.nbn, 5,
                                             (float)c.N);
  node_apply_kernel<<<ng, 256, 0, c.stream>>>(c.agg, c.nbn, wB, nullptr, n5, 1);
  run_conv(c, p3, wB);
  bn_finalize_kernel<<<1, 32, 0, c.stream>>>(c.nraw, F(86), F(85), c.nbn, 5,
                                             (float)c.N);
  node_apply_kernel<<<ng, 256, 0, c.stream>>>(c.agg, c.nbn, out, wA, n5, 1);
}

extern "C" void kernel_launch(void* const* d_in, const int* in_sizes, int n_in,
                              void* d_out, int out_size, void* d_ws,
                              size_t ws_size, hipStream_t stream) {
  const int N = in_sizes[0] / 5;
  const int E = in_sizes[2] / 11;
  const float* x = (const float*)d_in[0];
  const int* ei = (const int*)d_in[1];
  const float* ea = (const float*)d_in[2];
  auto F = [&](int i) { return (const float*)d_in[i]; };

  float* W = (float*)d_ws;
  const int NB = N * 5;
  Ctx c;
  c.stream = stream; c.E = E; c.N = N; c.ea = ea;
  c.src = ei; c.dst = ei + E;
  c.eaStats = W + 0;    c.eaRaw = W + 64;   c.rawAll = W + 128;
  c.bnBuf = W + 512;    c.nraw = W + 896;   c.nbn = W + 960;
  c.araw = W + 1024;    c.abn = W + 1088;
  float* nb[23];
  for (int i = 0; i < 23; ++i) nb[i] = W + 1280 + (size_t)i * NB;
  c.agg = nb[22];
  float* hcat = W + 1280 + (size_t)23 * NB;
  float* atts = hcat + (size_t)N * 90;
  float* t4 = atts + (size_t)N * 90;

  const int n5 = N * 5, ng = (n5 + 255) / 256, nn = (N + 255) / 256;

  // edge_attr global stats (shared by every conv's bn0)
  (void)hipMemsetAsync(c.eaRaw, 0, 64 * sizeof(float), stream);
  ea_stats_kernel<<<128, 256, 0, stream>>>(ea, c.eaRaw, E);
  ea_finalize_kernel<<<1, 32, 0, stream>>>(c.eaRaw, c.eaStats, E);

  // top-level x BN -> x0
  float* x0 = nb[21];
  (void)hipMemsetAsync(c.nraw, 0, 64 * sizeof(float), stream);
  node_stats_kernel<<<nn, 256, 0, stream>>>(x, c.nraw, N);
  bn_finalize_kernel<<<1, 32, 0, stream>>>(c.nraw, F(1580), F(1579), c.nbn, 5,
                                           (float)N);
  node_apply_kernel<<<ng, 256, 0, stream>>>(x, c.nbn, x0, nullptr, n5, 0);

  // cat slots: 0=h1 1..6=h2..h7 7..12=x1..x6 13..17=y1..y5
  float* wA = nb[18]; float* wB = nb[19]; float* cur = nb[20];
  float** cat = nb;
  run_block(c, d_in, 0, x0, cat[1], wA, wB);
  run_block(c, d_in, 1, cat[1], cat[7], wA, wB);
  run_block(c, d_in, 2, cat[7], cat[13], wA, wB);
  add3_kernel<<<ng, 256, 0, stream>>>(cur, cat[13], cat[1], x0, n5);
  run_block(c, d_in, 3, cur, cat[2], wA, wB);
  run_block(c, d_in, 4, cat[2], cat[8], wA, wB);
  run_block(c, d_in, 5, cat[8], cat[14], wA, wB);
  add3_kernel<<<ng, 256, 0, stream>>>(cur, cat[14], cat[1], cat[2], n5);
  run_block(c, d_in, 6, cur, cat[3], wA, wB);
  run_block(c, d_in, 7, cat[3], cat[9], wA, wB);
  run_block(c, d_in, 8, cat[9], cat[15], wA, wB);
  add3_kernel<<<ng, 256, 0, stream>>>(cur, cat[15], cat[2], cat[3], n5);
  run_block(c, d_in, 9, cur, cat[4], wA, wB);
  run_block(c, d_in, 10, cat[4], cat[10], wA, wB);
  run_block(c, d_in, 11, cat[10], cat[16], wA, wB);
  add3_kernel<<<ng, 256, 0, stream>>>(cur, cat[16], cat[3], cat[4], n5);
  run_block(c, d_in, 12, cur, cat[5], wA, wB);
  run_block(c, d_in, 13, cat[5], cat[11], wA, wB);
  run_block(c, d_in, 14, cat[11], cat[17], wA, wB);
  add3_kernel<<<ng, 256, 0, stream>>>(cur, cat[17], cat[4], cat[5], n5);
  run_block(c, d_in, 15, cur, cat[6], wA, wB);
  run_block(c, d_in, 16, cat[6], cat[12], wA, wB);
  run_block(c, d_in, 17, cat[12], cat[0], wA, wB);

  // concat -> [N,90]
  Cat18 cc;
  for (int i = 0; i < 18; ++i) cc.p[i] = cat[i];
  concat_kernel<<<nn, 256, 0, stream>>>(cc, hcat, N);

  // cross attention x2 (att1 leaves at 3..7, att2 at 8..12)
  (void)hipMemsetAsync(c.araw, 0, 64 * sizeof(float), stream);
  catt1_kernel<<<nn, 256, 0, stream>>>(hcat, F(6), F(5), t4, c.araw, N);
  bn_finalize_kernel<<<1, 32, 0, stream>>>(c.araw, F(4), F(3), c.abn, 4,
                                           (float)N);
  catt2_kernel<<<nn, 256, 0, stream>>>(hcat, t4, c.abn, F(7), atts, N, 0);
  (void)hipMemsetAsync(c.araw, 0, 64 * sizeof(float), stream);
  catt1_kernel<<<nn, 256, 0, stream>>>(hcat, F(11), F(10), t4, c.araw, N);
  bn_finalize_kernel<<<1, 32, 0, stream>>>(c.araw, F(9), F(8), c.abn, 4,
                                           (float)N);
  catt2_kernel<<<nn, 256, 0, stream>>>(hcat, t4, c.abn, F(12), atts, N, 1);

  // final linear 90 -> 1
  final_lin_kernel<<<nn, 256, 0, stream>>>(atts, F(1582), F(1581),
                                           (float*)d_out, N);
  (void)n_in; (void)out_size; (void)ws_size;
}
